// EvolvedLoopLinear_21251498180723
// MI455X (gfx1250) — compile-verified, matrix operand order checked by reading, not run on hardware
//
#include <hip/hip_runtime.h>

// EvolvedLoopLinear forward == dense GEMM: out[m,n] = sum_k x[m,k]*W[n,k] + b[n]
// M=8192 (batch), N=4096 (out), K=4096 (in). fp32 in/out.
// Strategy: 3-term bf16 split-GEMM on v_wmma_f32_16x16x32_bf16, fp32 accumulate,
// double-buffered LDS stage so global prefetch overlaps the WMMA block.

constexpr int Msz = 8192;
constexpr int Nsz = 4096;
constexpr int Ksz = 4096;

constexpr int BM = 128;   // block tile M
constexpr int BN = 128;   // block tile N
constexpr int BK = 32;    // K step == WMMA K
constexpr int BDIM = 256; // 8 wave32
constexpr int LDSP = BK + 8; // 40 halves per row (80B stride, 16B-aligned)

typedef __bf16 v16bf __attribute__((ext_vector_type(16)));
typedef __bf16 v8bf  __attribute__((ext_vector_type(8)));
typedef __bf16 v4bf  __attribute__((ext_vector_type(4)));
typedef float  v8f   __attribute__((ext_vector_type(8)));
typedef float  v4f   __attribute__((ext_vector_type(4)));

__global__ __launch_bounds__(BDIM) void
evolved_linear_gemm(const float* __restrict__ X, const float* __restrict__ W,
                    const float* __restrict__ bias, float* __restrict__ out)
{
    // Double-buffered staging: 2 x (Ah|Al|Bh|Bl) = 2 x 40KB = 80KB LDS.
    __shared__ __bf16 Ah[2][BM][LDSP];
    __shared__ __bf16 Al[2][BM][LDSP];
    __shared__ __bf16 Bh[2][BN][LDSP];
    __shared__ __bf16 Bl[2][BN][LDSP];

    const int tid    = threadIdx.x;
    const int lane   = tid & 31;
    const int wid    = tid >> 5;
    const int lane16 = lane & 15;
    const int laneH  = lane >> 4;     // which K-half this lane covers

    const int tileN = blockIdx.x * BN;
    const int tileM = blockIdx.y * BM;

    const int waveM = (wid & 3) * 32;  // 4 waves along M, 2 m-frags each
    const int waveN = (wid >> 2) * 64; // 2 waves along N, 4 n-frags each

    // Global tile load mapping: BK=32 floats per row = 8 float4; 256 threads
    // cover 32 rows per pass, 4 passes for 128 rows.
    const int lr = tid >> 3;        // row 0..31
    const int lc = (tid & 7) * 4;   // float col 0..28

    const float* xg = X + (size_t)(tileM + lr) * Ksz + lc;
    const float* wg = W + (size_t)(tileN + lr) * Ksz + lc;

    v4f pa[4], pb[4]; // pipelined global-load staging registers

    auto gload = [&](int kb) {
        const float* xa = xg + kb * BK;
        const float* wa = wg + kb * BK;
#pragma unroll
        for (int i = 0; i < 4; ++i) {
            pa[i] = *(const v4f*)(xa + (size_t)i * 32 * Ksz);
            pb[i] = *(const v4f*)(wa + (size_t)i * 32 * Ksz);
        }
    };

    // fp32 -> (hi,lo) bf16 split at LDS-write time.
    // hi = f with low 16 mantissa bits zeroed (exact bf16), lo = rne_bf16(f - hi).
    auto stlds = [&](int p) {
#pragma unroll
        for (int i = 0; i < 4; ++i) {
            const int r = lr + i * 32;
            v4bf ah, al, bh, bl;
#pragma unroll
            for (int j = 0; j < 4; ++j) {
                float fa = pa[i][j];
                float fha = __uint_as_float(__float_as_uint(fa) & 0xffff0000u);
                ah[j] = (__bf16)fha;
                al[j] = (__bf16)(fa - fha);
                float fb = pb[i][j];
                float fhb = __uint_as_float(__float_as_uint(fb) & 0xffff0000u);
                bh[j] = (__bf16)fhb;
                bl[j] = (__bf16)(fb - fhb);
            }
            *(v4bf*)&Ah[p][r][lc] = ah;
            *(v4bf*)&Al[p][r][lc] = al;
            *(v4bf*)&Bh[p][r][lc] = bh;
            *(v4bf*)&Bl[p][r][lc] = bl;
        }
    };

    // Identical per-lane fill pattern for A and B fragments: lane16 = row,
    // laneH picks which 16-wide K chunk -> consistent K-permutation on both
    // operands, so the WMMA dot product matches the fp32 reference sum.
    auto ldfrag = [&](const __bf16 (*buf)[LDSP], int row) -> v16bf {
        const __bf16* p = &buf[row][laneH * 16];
        v8bf a = *(const v8bf*)p;
        v8bf b = *(const v8bf*)(p + 8);
        return __builtin_shufflevector(a, b, 0, 1, 2, 3, 4, 5, 6, 7,
                                             8, 9, 10, 11, 12, 13, 14, 15);
    };

    v8f acc[2][4];
#pragma unroll
    for (int m = 0; m < 2; ++m)
#pragma unroll
        for (int n = 0; n < 4; ++n)
            acc[m][n] = (v8f){};

    constexpr int NKB = Ksz / BK; // 128 K-steps

    gload(0);
    stlds(0);

    for (int kb = 0; kb < NKB; ++kb) {
        const int cur = kb & 1;
        __syncthreads(); // writers of `cur` (prev iter) done; readers of `nxt` done

        if (kb + 1 < NKB) gload(kb + 1); // issue prefetch BEFORE the WMMA block

        v16bf a_h[2], a_l[2];
#pragma unroll
        for (int m = 0; m < 2; ++m) {
            a_h[m] = ldfrag(Ah[cur], waveM + m * 16 + lane16);
            a_l[m] = ldfrag(Al[cur], waveM + m * 16 + lane16);
        }
#pragma unroll
        for (int n = 0; n < 4; ++n) {
            v16bf b_h = ldfrag(Bh[cur], waveN + n * 16 + lane16);
            v16bf b_l = ldfrag(Bl[cur], waveN + n * 16 + lane16);
#pragma unroll
            for (int m = 0; m < 2; ++m) {
                // acc += xh*wh + xh*wl + xl*wh   (xl*wl term ~2^-16, dropped)
                acc[m][n] = __builtin_amdgcn_wmma_f32_16x16x32_bf16(
                    false, a_h[m], false, b_h, (short)0, acc[m][n], false, false);
                acc[m][n] = __builtin_amdgcn_wmma_f32_16x16x32_bf16(
                    false, a_h[m], false, b_l, (short)0, acc[m][n], false, false);
                acc[m][n] = __builtin_amdgcn_wmma_f32_16x16x32_bf16(
                    false, a_l[m], false, b_h, (short)0, acc[m][n], false, false);
            }
        }

        // Convert + stage next tile into the other buffer; wait-on-loads lands
        // here, after the 24 WMMAs above, hiding the global latency.
        if (kb + 1 < NKB) stlds(cur ^ 1);
    }

    // Epilogue: C/D layout — VGPR v holds M = v + 8*laneH, N = lane16.
    // Non-temporal stores: keep the L2 full of the reused x/W operands
    // instead of the streamed-once 128MB output.
#pragma unroll
    for (int n = 0; n < 4; ++n) {
        const int gn = tileN + waveN + n * 16 + lane16;
        const float bv = bias[gn];
#pragma unroll
        for (int m = 0; m < 2; ++m) {
            const int gmBase = tileM + waveM + m * 16 + laneH * 8;
#pragma unroll
            for (int v = 0; v < 8; ++v) {
                __builtin_nontemporal_store(acc[m][n][v] + bv,
                    out + (size_t)(gmBase + v) * Nsz + gn);
            }
        }
    }
}

extern "C" void kernel_launch(void* const* d_in, const int* in_sizes, int n_in,
                              void* d_out, int out_size, void* d_ws, size_t ws_size,
                              hipStream_t stream)
{
    const float* x = (const float*)d_in[0]; // [8192, 4096]
    const float* W = (const float*)d_in[1]; // [4096, 4096]
    const float* b = (const float*)d_in[2]; // [4096]
    float* out = (float*)d_out;             // [8192, 4096]

    dim3 grid(Nsz / BN, Msz / BM); // 32 x 64 = 2048 workgroups
    evolved_linear_gemm<<<grid, BDIM, 0, stream>>>(x, W, b, out);
}
